// OperationEmbeddingLayer_1717986918539
// MI455X (gfx1250) — matrix operand order repaired and checked by reading, not compiled, not run on hardware
//
#include <hip/hip_runtime.h>
#include <hip/hip_bf16.h>

typedef __attribute__((ext_vector_type(16))) __bf16 v16bf;
typedef __attribute__((ext_vector_type(8)))  __bf16 v8bf;
typedef __attribute__((ext_vector_type(8)))  float  v8f;
typedef __attribute__((ext_vector_type(4)))  float  v4f;

#define NOPS 200000
#define HID  256
#define OUTD 128
#define ACT_STRIDE 264   // 256 + 8 bf16 pad -> row stride 132 dwords, spreads LDS banks

// ---------------------------------------------------------------- utilities

__global__ __launch_bounds__(256) void zero_f32_kernel(float* __restrict__ p, size_t n) {
    size_t i = (size_t)blockIdx.x * blockDim.x + threadIdx.x;
    if (i < n) p[i] = 0.0f;
}

// fp32 -> bf16 (RNE) with K-padding to Kpad (zero fill)
__global__ __launch_bounds__(256) void cvt_bf16_pad_kernel(const float* __restrict__ src,
                                                           unsigned short* __restrict__ dst,
                                                           int N, int K, int Kpad) {
    int i = blockIdx.x * blockDim.x + threadIdx.x;
    if (i >= N * Kpad) return;
    int n = i / Kpad, k = i % Kpad;
    float v = (k < K) ? src[(size_t)n * K + k] : 0.0f;
    unsigned u = __builtin_bit_cast(unsigned, v);
    unsigned r = (u + 0x7FFFu + ((u >> 16) & 1u)) >> 16;
    dst[i] = (unsigned short)r;
}

// edge-parallel scatter-add: out[dst[e]][f] += feat[src[e]][f]
template <int D>
__global__ __launch_bounds__(256) void scatter_add_kernel(const float* __restrict__ feat,
                                                          const int* __restrict__ srcIdx,
                                                          const int* __restrict__ dstIdx,
                                                          float* __restrict__ out, int nEdges) {
    size_t i = (size_t)blockIdx.x * blockDim.x + threadIdx.x;
    if (i >= (size_t)nEdges * D) return;
    int e = (int)(i / D);
    int f = (int)(i % D);
    float v = feat[(size_t)srcIdx[e] * D + f];
    __hip_atomic_fetch_add(out + (size_t)dstIdx[e] * D + f, v,
                           __ATOMIC_RELAXED, __HIP_MEMORY_SCOPE_AGENT);
}

// ---------------------------------------------------------------- fused 3-layer MLP

__device__ __forceinline__ v8f wmma_bf16(v16bf a, v16bf b, v8f c) {
    return __builtin_amdgcn_wmma_f32_16x16x32_bf16(false, a, false, b, (short)0, c, false, false);
}

__device__ __forceinline__ float elu(float v) {
    return v > 0.0f ? v : (__expf(v) - 1.0f);
}

// One block = 16 rows through  d_in -> 256 (ELU) -> 256 (ELU) -> 128.
// 8 waves: waves own 2 N-tiles for the 256-wide layers, 1 N-tile for the 128-wide layer.
template <int D_IN, bool GATHER, bool IN_BF16, bool OUT_F32>
__global__ __launch_bounds__(256)
void mlp3_kernel(const void* __restrict__ inp, const int* __restrict__ gidx,
                 const unsigned short* __restrict__ W1p, const float* __restrict__ B1,
                 const unsigned short* __restrict__ W2p, const float* __restrict__ B2,
                 const unsigned short* __restrict__ W3p, const float* __restrict__ B3,
                 void* __restrict__ outp, int outStride, int colOff) {
    constexpr int K1PAD = (D_IN + 31) & ~31;

    __shared__ __bf16 act1[16 * ACT_STRIDE];
    __shared__ __bf16 act2[16 * ACT_STRIDE];

    const int tid  = threadIdx.x;
    const int wave = tid >> 5;
    const int lane = tid & 31;
    const int lr   = lane & 15;   // A-row / B-col / D-col within tile
    const int lh   = lane >> 4;   // half-wave selector
    const int row0 = blockIdx.x * 16;

    const __bf16* W1 = (const __bf16*)W1p;
    const __bf16* W2 = (const __bf16*)W2p;
    const __bf16* W3 = (const __bf16*)W3p;

    // layer-1 source row for this lane (lanes lr and lr+16 share row lr)
    const int rowA = row0 + lr;
    const float*  arow_f = nullptr;
    const __bf16* arow_b = nullptr;
    if constexpr (IN_BF16) {
        arow_b = (const __bf16*)inp + (size_t)rowA * D_IN;
    } else if constexpr (GATHER) {
        arow_f = (const float*)inp + (size_t)gidx[rowA] * D_IN;
    } else {
        arow_f = (const float*)inp + (size_t)rowA * D_IN;
    }

    // A fragment from global (contiguous k: k0 + (l/16)*16 .. +16, matching B)
    auto loadA1 = [&](int k0) -> v16bf {
        v16bf a;
        const int c0 = k0 + lh * 16;
#pragma unroll
        for (int h = 0; h < 2; ++h) {
            const int c = c0 + h * 8;
            if (c < D_IN) {
                if constexpr (IN_BF16) {
                    v8bf x = *(const v8bf*)(arow_b + c);
#pragma unroll
                    for (int j = 0; j < 8; ++j) a[h * 8 + j] = x[j];
                } else {
                    v4f x0 = *(const v4f*)(arow_f + c);
                    v4f x1 = *(const v4f*)(arow_f + c + 4);
#pragma unroll
                    for (int j = 0; j < 4; ++j) {
                        a[h * 8 + j]     = (__bf16)x0[j];
                        a[h * 8 + 4 + j] = (__bf16)x1[j];
                    }
                }
            } else {
#pragma unroll
                for (int j = 0; j < 8; ++j) a[h * 8 + j] = (__bf16)0.0f;
            }
        }
        return a;
    };

    // A fragment from LDS activations
    auto loadA_lds = [&](const __bf16* actB, int k0) -> v16bf {
        const __bf16* p = actB + lr * ACT_STRIDE + k0 + lh * 16;
        v8bf x0 = *(const v8bf*)p;
        v8bf x1 = *(const v8bf*)(p + 8);
        v16bf a;
#pragma unroll
        for (int j = 0; j < 8; ++j) { a[j] = x0[j]; a[8 + j] = x1[j]; }
        return a;
    };

    // B fragment: lane reads W[n][k0 + (l/16)*16 .. +16] contiguous (B[k][n] = W[n][k])
    auto loadB = [&](const __bf16* W, int Kpad, int n, int k0) -> v16bf {
        return *(const v16bf*)(W + (size_t)n * Kpad + k0 + lh * 16);
    };

    // D tile -> LDS activation with bias + ELU (D layout: vgpr r -> row r+8*lh, col lr)
    auto storeAct = [&](__bf16* actB, v8f acc, float bias, int nTile) {
#pragma unroll
        for (int r = 0; r < 8; ++r) {
            float v = elu(acc[r] + bias);
            actB[(r + lh * 8) * ACT_STRIDE + nTile * 16 + lr] = (__bf16)v;
        }
    };

    // ---- layer 1: D_IN -> 256 ----
    {
        const int t0 = wave * 2, t1 = wave * 2 + 1;
        const int n0 = t0 * 16, n1 = t1 * 16;
        v8f acc0 = {0, 0, 0, 0, 0, 0, 0, 0};
        v8f acc1 = {0, 0, 0, 0, 0, 0, 0, 0};
        for (int k0 = 0; k0 < K1PAD; k0 += 32) {
            v16bf a  = loadA1(k0);
            v16bf b0 = loadB(W1, K1PAD, n0 + lr, k0);
            v16bf b1 = loadB(W1, K1PAD, n1 + lr, k0);
            acc0 = wmma_bf16(a, b0, acc0);
            acc1 = wmma_bf16(a, b1, acc1);
        }
        storeAct(act1, acc0, B1[n0 + lr], t0);
        storeAct(act1, acc1, B1[n1 + lr], t1);
    }
    __syncthreads();

    // ---- layer 2: 256 -> 256 ----
    {
        const int t0 = wave * 2, t1 = wave * 2 + 1;
        const int n0 = t0 * 16, n1 = t1 * 16;
        v8f acc0 = {0, 0, 0, 0, 0, 0, 0, 0};
        v8f acc1 = {0, 0, 0, 0, 0, 0, 0, 0};
        for (int k0 = 0; k0 < HID; k0 += 32) {
            v16bf a  = loadA_lds(act1, k0);
            v16bf b0 = loadB(W2, HID, n0 + lr, k0);
            v16bf b1 = loadB(W2, HID, n1 + lr, k0);
            acc0 = wmma_bf16(a, b0, acc0);
            acc1 = wmma_bf16(a, b1, acc1);
        }
        storeAct(act2, acc0, B2[n0 + lr], t0);
        storeAct(act2, acc1, B2[n1 + lr], t1);
    }
    __syncthreads();

    // ---- layer 3: 256 -> 128 (no activation) ----
    {
        const int n0 = wave * 16;
        v8f acc = {0, 0, 0, 0, 0, 0, 0, 0};
        for (int k0 = 0; k0 < HID; k0 += 32) {
            v16bf a = loadA_lds(act2, k0);
            v16bf b = loadB(W3, HID, n0 + lr, k0);
            acc = wmma_bf16(a, b, acc);
        }
        const float bias = B3[n0 + lr];
#pragma unroll
        for (int r = 0; r < 8; ++r) {
            float v   = acc[r] + bias;
            int   row = row0 + r + lh * 8;
            int   col = colOff + n0 + lr;
            if constexpr (OUT_F32) {
                ((float*)outp)[(size_t)row * outStride + col] = v;
            } else {
                ((__bf16*)outp)[(size_t)row * outStride + col] = (__bf16)v;
            }
        }
    }
}

// ---------------------------------------------------------------- host

extern "C" void kernel_launch(void* const* d_in, const int* in_sizes, int n_in,
                              void* d_out, int out_size, void* d_ws, size_t ws_size,
                              hipStream_t stream) {
    const float* operations    = (const float*)d_in[0];
    const float* items         = (const float*)d_in[1];
    const int*   related_items = (const int*)d_in[2];
    const float* materials     = (const float*)d_in[3];
    const float* resources     = (const float*)d_in[4];
    const int*   mat_edge_op   = (const int*)d_in[5];
    const int*   mat_edge_mat  = (const int*)d_in[6];
    const int*   res_edge_op   = (const int*)d_in[7];
    const int*   res_edge_res  = (const int*)d_in[8];
    const int*   prec_op       = (const int*)d_in[9];
    const int*   prec_pred     = (const int*)d_in[10];

    // params dict insertion order: combined, items, predecessors, successors,
    // resources, materials, self; each tuple -> (W1,b1,W2,b2,W3,b3)
    auto P = [&](int branch, int leaf) -> const float* {
        return (const float*)d_in[11 + branch * 6 + leaf];
    };
    const int din[7] = {768, 64, 128, 128, 32, 16, 128};

    char*  ws  = (char*)d_ws;
    size_t off = 0;
    auto alloc = [&](size_t bytes) -> void* {
        void* p = (void*)(ws + off);
        off += (bytes + 255) & ~(size_t)255;
        return p;
    };

    // contiguous segment-sum region (one zero pass): pred | succ | res | mat
    float* seg      = (float*)alloc((size_t)NOPS * 304 * sizeof(float));
    float* seg_pred = seg;
    float* seg_succ = seg + (size_t)NOPS * 128;
    float* seg_res  = seg + (size_t)NOPS * 256;
    float* seg_mat  = seg + (size_t)NOPS * 288;
    unsigned short* concat = (unsigned short*)alloc((size_t)NOPS * 768 * 2);

    unsigned short *w1[7], *w2[7], *w3[7];
    for (int b = 0; b < 7; ++b) {
        int kp = (din[b] + 31) & ~31;
        w1[b] = (unsigned short*)alloc((size_t)256 * kp * 2);
        w2[b] = (unsigned short*)alloc((size_t)256 * 256 * 2);
        w3[b] = (unsigned short*)alloc((size_t)128 * 256 * 2);
    }

    // 1) zero segment sums
    {
        size_t n = (size_t)NOPS * 304;
        zero_f32_kernel<<<(int)((n + 255) / 256), 256, 0, stream>>>(seg, n);
    }

    // 2) convert all weights to bf16 (K-padded)
    auto cvt = [&](const float* src, unsigned short* dst, int N, int K, int Kpad) {
        int n = N * Kpad;
        cvt_bf16_pad_kernel<<<(n + 255) / 256, 256, 0, stream>>>(src, dst, N, K, Kpad);
    };
    for (int b = 0; b < 7; ++b) {
        int kp = (din[b] + 31) & ~31;
        cvt(P(b, 0), w1[b], 256, din[b], kp);
        cvt(P(b, 2), w2[b], 256, 256, 256);
        cvt(P(b, 4), w3[b], 128, 256, 256);
    }

    // 3) scatter-adds
    scatter_add_kernel<16><<<(800000 * 16 + 255) / 256, 256, 0, stream>>>(
        materials, mat_edge_mat, mat_edge_op, seg_mat, 800000);
    scatter_add_kernel<32><<<(int)((1600000ull * 32 + 255) / 256), 256, 0, stream>>>(
        resources, res_edge_res, res_edge_op, seg_res, 1600000);
    scatter_add_kernel<128><<<(int)((400000ull * 128 + 255) / 256), 256, 0, stream>>>(
        operations, prec_pred, prec_op, seg_pred, 400000);
    scatter_add_kernel<128><<<(int)((400000ull * 128 + 255) / 256), 256, 0, stream>>>(
        operations, prec_op, prec_pred, seg_succ, 400000);

    // 4) branch MLPs -> bf16 concat [pred|succ|res|mat|item|self]
    dim3 g(NOPS / 16), blk(256);
    mlp3_kernel<128, false, false, false><<<g, blk, 0, stream>>>(
        seg_pred, nullptr, w1[2], P(2, 1), w2[2], P(2, 3), w3[2], P(2, 5), concat, 768, 0);
    mlp3_kernel<128, false, false, false><<<g, blk, 0, stream>>>(
        seg_succ, nullptr, w1[3], P(3, 1), w2[3], P(3, 3), w3[3], P(3, 5), concat, 768, 128);
    mlp3_kernel<32, false, false, false><<<g, blk, 0, stream>>>(
        seg_res, nullptr, w1[4], P(4, 1), w2[4], P(4, 3), w3[4], P(4, 5), concat, 768, 256);
    mlp3_kernel<16, false, false, false><<<g, blk, 0, stream>>>(
        seg_mat, nullptr, w1[5], P(5, 1), w2[5], P(5, 3), w3[5], P(5, 5), concat, 768, 384);
    mlp3_kernel<64, true, false, false><<<g, blk, 0, stream>>>(
        items, related_items, w1[1], P(1, 1), w2[1], P(1, 3), w3[1], P(1, 5), concat, 768, 512);
    mlp3_kernel<128, false, false, false><<<g, blk, 0, stream>>>(
        operations, nullptr, w1[6], P(6, 1), w2[6], P(6, 3), w3[6], P(6, 5), concat, 768, 640);

    // 5) combined MLP: bf16 concat -> fp32 output
    mlp3_kernel<768, false, true, true><<<g, blk, 0, stream>>>(
        concat, nullptr, w1[0], P(0, 1), w2[0], P(0, 3), w3[0], P(0, 5), d_out, 128, 0);
}
